// SimpleOutlookAttention_79809082294677
// MI455X (gfx1250) — compile-verified, matrix operand order checked by reading, not run on hardware
//
#include <hip/hip_runtime.h>

// ---------------------------------------------------------------------------
// Fused Outlook Attention for MI455X (gfx1250, wave32, WMMA).
// One workgroup = (b, hh, 16 ww-sites) = 3x48 pixel patch. All intermediates
// (v, attn logits, softmax, pre-projection embeddings) stay in LDS; HBM
// traffic is read-x-once + write-out-once (~400 MB) -> memory bound at
// 23.3 TB/s (~17 us floor). All three GEMMs run on v_wmma_f32_16x16x32_bf16.
// v_w / out_w are staged once per workgroup into LDS in fragment-major bf16
// order (aliasing dead regions), so the GEMM inner loop is pure
// ds_load_b128 + v_wmma, and L2 weight traffic is read-once per workgroup.
// ---------------------------------------------------------------------------

typedef __attribute__((ext_vector_type(8)))  float  f32x8;
typedef __attribute__((ext_vector_type(4)))  __bf16 bf16x4;
typedef __attribute__((ext_vector_type(8)))  __bf16 bf16x8;
typedef __attribute__((ext_vector_type(16))) __bf16 bf16x16;

#define WG_THREADS 256
#define BATCH   8
#define HIMG    252
#define WIMG    192
#define CIN     64
#define HEADS   6
#define EMBED   192
#define KDIM    32
#define TW      16            // ww sites per workgroup
#define PIXW    48            // 3*TW pixels wide
#define NPIX    144           // 3*PIXW rows (A / E rows)
#define NATT    486           // K^4 * HEADS
#define NATT_T  31            // ceil(486/16) N-tiles for attn GEMM
#define ASTR    72            // bf16 A row stride (padded: 64+8)
#define VSTR    192           // fp32 V row stride
#define ESTR    200           // bf16 E row stride (padded: 192+8)
#define PSTR    72            // bf16 pooled row stride (padded)
#define SCALE_F 0.17677669529663687f   // 32^-0.5

// LDS layout (bytes), total 226432 <= 320KB WGP budget.
// Aliases: v_w fragments live in the E region (E dead until Phase 5);
//          out_w fragments live in the V region (V dead after Phase 5).
#define OFF_ABF   0            // 144*72*2  = 20736
#define OFF_V     20736        // 144*192*4 = 110592  (also out_w frags 73728)
#define OFF_E     131328       // 144*200*2 = 57600   (also v_w frags 24576)
#define OFF_POOL  188928       // 16*64*4   = 4096
#define OFF_ATTN  193024       // 16*486*4  = 31104
#define OFF_PBF   224128       // 16*72*2   = 2304
#define SMEM_BYTES 226432

// Native f32 -> bf16 (backend emits v_cvt_pk_bf16_f32, RNE)
static __device__ inline __bf16 f2bf(float f) { return (__bf16)f; }

// 16-bit A-fragment (16x32) from LDS. Lane holds row m, khalf=lane>>4:
// elements 0..7 -> K = k0+kh*8..+7, elements 8..15 -> K = k0+16+kh*8..+7.
static __device__ inline bf16x16 load_afrag_lds(const __bf16* A, int lda,
                                                int m, int k0, int kh) {
  const __bf16* p = A + m * lda + k0 + kh * 8;
  bf16x8 lo = *(const bf16x8*)p;          // ds_load_b128
  bf16x8 hi = *(const bf16x8*)(p + 16);   // ds_load_b128
  bf16x16 a;
#pragma unroll
  for (int e = 0; e < 8; ++e) { a[e] = lo[e]; a[8 + e] = hi[e]; }
  return a;
}

// Fragment-major staged B: frag f, lane -> 16 contiguous bf16 (32B).
static __device__ inline bf16x16 load_bfrag_lds(const __bf16* buf,
                                                int f, int lane) {
  return *(const bf16x16*)(buf + ((size_t)f * 32 + lane) * 16);
}

// Stage row-major global W[k][n] (ldw cols) into fragment-major bf16 LDS.
// Each weight element is read exactly once per workgroup (L2-hot).
static __device__ inline void stage_bfrags(const float* __restrict__ W, int ldw,
                                           int ntiles, int ksteps,
                                           __bf16* __restrict__ dst, int tid) {
  for (int t = tid; t < ntiles * ksteps * 32; t += WG_THREADS) {
    int lane = t & 31;
    int f    = t >> 5;
    int nt   = f / ksteps;
    int ks   = f - nt * ksteps;
    int n    = nt * 16 + (lane & 15);
    int kh   = lane >> 4;
    const float* p = W + (size_t)(ks * 32 + kh * 8) * ldw + n;
    bf16x16 bfr;
#pragma unroll
    for (int e = 0; e < 8; ++e) bfr[e] = f2bf(p[(size_t)e * ldw]);
    p += (size_t)16 * ldw;
#pragma unroll
    for (int e = 0; e < 8; ++e) bfr[8 + e] = f2bf(p[(size_t)e * ldw]);
    *(bf16x16*)(dst + (size_t)t * 16) = bfr;   // 32B contiguous ds store
  }
}

// Guarded global B-fragment for the N-padded attn GEMM (~1x read, no staging).
static __device__ inline bf16x16 load_bfrag_glb_guard(const float* __restrict__ W,
                                                      int ldw, int ncols,
                                                      int k0, int n, int kh) {
  const bool ok = (n < ncols);
  const float* p = W + (size_t)(k0 + kh * 8) * ldw + (ok ? n : 0);
  bf16x16 b;
#pragma unroll
  for (int e = 0; e < 8; ++e) b[e] = f2bf(ok ? p[(size_t)e * ldw] : 0.0f);
  p += (size_t)16 * ldw;
#pragma unroll
  for (int e = 0; e < 8; ++e) b[8 + e] = f2bf(ok ? p[(size_t)e * ldw] : 0.0f);
  return b;
}

__global__ __launch_bounds__(WG_THREADS, 1)
void outlook_fused_kernel(const float* __restrict__ x,
                          const float* __restrict__ v_w,
                          const float* __restrict__ attn_w,
                          const float* __restrict__ attn_b,
                          const float* __restrict__ out_w,
                          const float* __restrict__ out_b,
                          float* __restrict__ out) {
  extern __shared__ unsigned char smem[];
  __bf16* Abf      = (__bf16*)(smem + OFF_ABF);   // [144][72]  bf16 x-patch rows
  float*  V        = (float*)(smem + OFF_V);      // [144][192] fp32 v
  __bf16* E        = (__bf16*)(smem + OFF_E);     // [144][200] bf16 embeddings
  float*  pooled   = (float*)(smem + OFF_POOL);   // [16][64]   fp32 pool acc
  float*  attnL    = (float*)(smem + OFF_ATTN);   // [16][486]
  __bf16* pooledBf = (__bf16*)(smem + OFF_PBF);   // [16][72]   bf16 pooled tile
  __bf16* vwFrag   = (__bf16*)(smem + OFF_E);     // alias: v_w frags (Ph1-2b)
  __bf16* owFrag   = (__bf16*)(smem + OFF_V);     // alias: out_w frags (Ph6)

  const int tid  = threadIdx.x;
  const int wave = tid >> 5;
  const int lane = tid & 31;
  const int nl   = lane & 15;   // column within 16-wide tile
  const int kh   = lane >> 4;   // khalf (frag load) / row-half (C/D)

  const int b  = blockIdx.z;
  const int hh = blockIdx.y;
  const int w0 = blockIdx.x * PIXW;

  // ---- Phase 0: zero pooled accumulators --------------------------------
  for (int i = tid; i < TW * CIN; i += WG_THREADS) pooled[i] = 0.0f;
  __syncthreads();

  // ---- Phase 1a: stage v_w fragments into (dead) E region ---------------
  stage_bfrags(v_w, EMBED, 12, 2, vwFrag, tid);

  // ---- Phase 1b: x patch load (float4, read-once), bf16 A rows,
  //                3x3 mean-pool via ds_add_f32 ---------------------------
  {
    const float inv9 = 1.0f / 9.0f;
    for (int idx = tid; idx < (NPIX * CIN) / 4; idx += WG_THREADS) {
      int pix = idx >> 4;               // /16 float4 chunks per pixel
      int c4  = (idx & 15) * 4;
      int pr  = pix / PIXW, pc = pix % PIXW;
      int h   = hh * 3 + pr, w = w0 + pc;
      const float4 xv =
          *(const float4*)&x[(((size_t)b * HIMG + h) * WIMG + w) * CIN + c4];
      int site = pc / 3, kc = pc % 3;
      int rv   = site * 9 + pr * 3 + kc;          // v-row ordering
      bf16x4 bx = { f2bf(xv.x), f2bf(xv.y), f2bf(xv.z), f2bf(xv.w) };
      *(bf16x4*)(Abf + rv * ASTR + c4) = bx;      // ds_store_b64
      float* pp = pooled + site * CIN + c4;
      atomicAdd(pp + 0, xv.x * inv9);
      atomicAdd(pp + 1, xv.y * inv9);
      atomicAdd(pp + 2, xv.z * inv9);
      atomicAdd(pp + 3, xv.w * inv9);
    }
  }
  __syncthreads();

  // ---- Phase 2a: stage pooled as bf16 A-tile ----------------------------
  for (int i = tid; i < TW * CIN; i += WG_THREADS) {
    int site = i >> 6, c = i & 63;
    pooledBf[site * PSTR + c] = f2bf(pooled[i]);
  }

  // ---- Phase 2b: v = A(144x64) @ v_w(64x192) via WMMA (all-LDS frags) ---
  for (int job = wave; job < 9 * 12; job += 8) {
    int mt = job / 12, nt = job % 12;
    f32x8 acc = {};
#pragma unroll
    for (int ks = 0; ks < 2; ++ks) {
      bf16x16 a  = load_afrag_lds(Abf, ASTR, mt * 16 + nl, ks * 32, kh);
      bf16x16 bb = load_bfrag_lds(vwFrag, nt * 2 + ks, lane);
      acc = __builtin_amdgcn_wmma_f32_16x16x32_bf16(
          false, a, false, bb, (short)0, acc, false, false);
    }
    float* vp = V + (size_t)(mt * 16 + 8 * kh) * VSTR + nt * 16 + nl;
#pragma unroll
    for (int r = 0; r < 8; ++r) vp[(size_t)r * VSTR] = acc[r];
  }
  __syncthreads();

  // ---- Phase 3: attn = (pooled @ attn_w + attn_b) * SCALE via WMMA ------
  //      M=16 (one tile), N=486 padded to 31x16, K=64; attn_w ~read-once.
  for (int nt = wave; nt < NATT_T; nt += 8) {
    f32x8 acc = {};
#pragma unroll
    for (int ks = 0; ks < 2; ++ks) {
      bf16x16 a  = load_afrag_lds(pooledBf, PSTR, nl, ks * 32, kh);
      bf16x16 bb = load_bfrag_glb_guard(attn_w, NATT, NATT,
                                        ks * 32, nt * 16 + nl, kh);
      acc = __builtin_amdgcn_wmma_f32_16x16x32_bf16(
          false, a, false, bb, (short)0, acc, false, false);
    }
    int col = nt * 16 + nl;
    if (col < NATT) {
      float bias = attn_b[col];
#pragma unroll
      for (int r = 0; r < 8; ++r) {
        int site = 8 * kh + r;                     // D row = site (M=16)
        attnL[site * NATT + col] = (acc[r] + bias) * SCALE_F;
      }
    }
  }
  __syncthreads();

  // ---- Phase 4: softmax over last dim (rows of 9) ------------------------
  for (int r = tid; r < TW * HEADS * 9; r += WG_THREADS) {
    int site = r / 54, rem = r % 54;       // rem = head*9 + p
    float* row = attnL + site * NATT + rem * 9;
    float m = row[0];
#pragma unroll
    for (int q = 1; q < 9; ++q) m = fmaxf(m, row[q]);
    float s = 0.0f;
#pragma unroll
    for (int q = 0; q < 9; ++q) { float e = __expf(row[q] - m); row[q] = e; s += e; }
    float inv = 1.0f / s;
#pragma unroll
    for (int q = 0; q < 9; ++q) row[q] *= inv;
  }
  __syncthreads();

  // ---- Phase 5: attn @ v (K=9, VALU) -> E in bf16, permuted to pixels ----
  //      (overwrites the dead v_w fragment alias)
  for (int idx = tid; idx < TW * HEADS * 9 * KDIM; idx += WG_THREADS) {
    int d    = idx & 31;
    int p    = (idx >> 5) % 9;
    int head = (idx / 288) % HEADS;
    int site = idx / (288 * HEADS);
    const float* arow = attnL + site * NATT + head * 81 + p * 9;
    const float* vcol = V + (size_t)(site * 9) * VSTR + head * KDIM + d;
    float acc = 0.0f;
#pragma unroll
    for (int q = 0; q < 9; ++q) acc = fmaf(arow[q], vcol[(size_t)q * VSTR], acc);
    int kr = p / 3, kc = p % 3;
    int erow = kr * PIXW + site * 3 + kc;          // output pixel ordering
    E[erow * ESTR + head * KDIM + d] = f2bf(acc);
  }
  __syncthreads();

  // ---- Phase 6a: stage out_w fragments into (now dead) V region ---------
  stage_bfrags(out_w, EMBED, 12, 6, owFrag, tid);
  __syncthreads();

  // ---- Phase 6b: final projection E(144x192) @ out_w(192x192) + bias ----
  for (int job = wave; job < 9 * 12; job += 8) {
    int mt = job / 12, nt = job % 12;
    f32x8 acc = {};
#pragma unroll
    for (int ks = 0; ks < 6; ++ks) {
      bf16x16 a  = load_afrag_lds(E, ESTR, mt * 16 + nl, ks * 32, kh);
      bf16x16 bb = load_bfrag_lds(owFrag, nt * 6 + ks, lane);
      acc = __builtin_amdgcn_wmma_f32_16x16x32_bf16(
          false, a, false, bb, (short)0, acc, false, false);
    }
    int col = nt * 16 + nl;
    float bias = out_b[col];
#pragma unroll
    for (int r = 0; r < 8; ++r) {
      int p  = mt * 16 + 8 * kh + r;
      int pr = p / PIXW, pc = p % PIXW;
      int h  = hh * 3 + pr, w = w0 + pc;
      out[(((size_t)b * HIMG + h) * WIMG + w) * EMBED + col] = acc[r] + bias;
    }
  }
}

extern "C" void kernel_launch(void* const* d_in, const int* in_sizes, int n_in,
                              void* d_out, int out_size, void* d_ws, size_t ws_size,
                              hipStream_t stream) {
  (void)in_sizes; (void)n_in; (void)d_ws; (void)ws_size; (void)out_size;
  const float* x      = (const float*)d_in[0];
  const float* v_w    = (const float*)d_in[1];
  const float* attn_w = (const float*)d_in[2];
  const float* attn_b = (const float*)d_in[3];
  const float* out_w  = (const float*)d_in[4];
  const float* out_b  = (const float*)d_in[5];
  float* out = (float*)d_out;

  // Idempotent + deterministic (no guards, per harness contract).
  hipFuncSetAttribute((const void*)outlook_fused_kernel,
                      hipFuncAttributeMaxDynamicSharedMemorySize, SMEM_BYTES);

  dim3 grid(64 / TW, 84, BATCH);   // (ww tiles, hh, batch) = (4, 84, 8)
  dim3 block(WG_THREADS);
  outlook_fused_kernel<<<grid, block, SMEM_BYTES, stream>>>(
      x, v_w, attn_w, attn_b, out_w, out_b, out);
}
